// C2fBoT_81063212745480
// MI455X (gfx1250) — matrix operand order repaired and checked by reading, not compiled, not run on hardware
//
#include <hip/hip_runtime.h>
#include <hip/hip_bf16.h>

typedef __attribute__((ext_vector_type(16))) _Float16 v16h;
typedef __attribute__((ext_vector_type(8)))  float    v8f;
typedef __attribute__((ext_vector_type(4)))  int      i32x4;

union AF { v16h v; uint4 q[2]; };

#define NTOT 8192   // B * H * W = 8 * 1024 columns
#define BN_EPS 1e-3f

__device__ __forceinline__ v8f wmma16(v16h a, v16h b, v8f c) {
  // D = A(16x32 f16) * B(32x16 f16) + C(16x16 f32)
  return __builtin_amdgcn_wmma_f32_16x16x32_f16(false, a, false, b, (short)0, c,
                                                false, false);
}

__device__ __forceinline__ float silu_f(float x) {
  return x / (1.0f + __expf(-x));
}

// ---- async global->LDS 16B copy (GLOBAL_LOAD_ASYNC_TO_LDS_B128, ASYNCcnt) --
#if __has_builtin(__builtin_amdgcn_global_load_async_to_lds_b128)
#define HAVE_ASYNC_LDS 1
#else
#define HAVE_ASYNC_LDS 0
#endif

__device__ __forceinline__ void lds_copy16_async(const _Float16* g,
                                                 _Float16* l) {
#if HAVE_ASYNC_LDS
  __builtin_amdgcn_global_load_async_to_lds_b128(
      (__attribute__((address_space(1))) i32x4*)g,
      (__attribute__((address_space(3))) i32x4*)l, 0, 0);
#else
  *(uint4*)l = *(const uint4*)g;
#endif
}

__device__ __forceinline__ void async_wait0() {
#if HAVE_ASYNC_LDS
#if __has_builtin(__builtin_amdgcn_s_wait_asynccnt)
  __builtin_amdgcn_s_wait_asynccnt(0);
#else
  asm volatile("s_wait_asynccnt 0x0" ::: "memory");
#endif
#endif
}

// ---------------------------------------------------------------------------
// Convert input x (NCHW fp32) -> X0 f16 matrix [512][8192], col n = b*1024+p
// ---------------------------------------------------------------------------
__global__ __launch_bounds__(256) void k_x_to_f16(const float* __restrict__ x,
                                                  _Float16* __restrict__ X0) {
  int idx = blockIdx.x * 256 + threadIdx.x;       // c*8192 + n
  if (idx >= 512 * NTOT) return;
  int c = idx >> 13, n = idx & 8191, b = n >> 10, p = n & 1023;
  X0[idx] = (_Float16)x[((b << 9) + c) * 1024 + p];
}

// ---------------------------------------------------------------------------
// Fold BN scale into weights (fp32 -> f16), emit per-channel bias.
// bn layout: [g | b | m | v] each Cout floats.  bn == nullptr -> identity.
// ---------------------------------------------------------------------------
__global__ __launch_bounds__(256) void k_fold_w(const float* __restrict__ w,
                                                const float* __restrict__ bn,
                                                int Cout, int K,
                                                _Float16* __restrict__ wo,
                                                float* __restrict__ bias) {
  int idx = blockIdx.x * 256 + threadIdx.x;
  if (idx >= Cout * K) return;
  int co = idx / K, k = idx - co * K;
  float scale = 1.0f, bia = 0.0f;
  if (bn) {
    float g = bn[co], bb = bn[Cout + co], m = bn[2 * Cout + co],
          v = bn[3 * Cout + co];
    scale = g * __frsqrt_rn(v + BN_EPS);
    bia = bb - m * scale;
  }
  wo[idx] = (_Float16)(w[idx] * scale);
  if (k == 0) bias[co] = bia;
}

// ---------------------------------------------------------------------------
// WMMA GEMM:  Out[m][n] = epi( sum_k W[m][k] * B(k,n) + bias[m] )
//   N = 8192 fixed.  Block tile 64(M) x 128(N), BK = 32, 256 threads = 8 waves
//   wave grid 4(M) x 2(N); each wave: 1 A-frag reused over 4 B-frags / k-step.
//   A tile staged via GLOBAL_LOAD_ASYNC_TO_LDS_B128 (ASYNCcnt path).
//   mode 0: B(k,n) = Bmat[k][n]           (1x1 conv)
//   mode 1: implicit im2col of 3x3/pad1:  k = ci*9 + ky*3 + kx over Bmat[256][8192]
//   epilogue: optional SiLU, optional residual (f16 matrix), output either
//   f16 matrix (Out) or fp32 NCHW (OutF32, Mdim out-channels).
// ---------------------------------------------------------------------------
__global__ __launch_bounds__(256) void k_gemm(
    const _Float16* __restrict__ W, const float* __restrict__ bias,
    const _Float16* __restrict__ Bmat, _Float16* __restrict__ Out,
    float* __restrict__ OutF32, const _Float16* __restrict__ Res,
    int Mdim, int Kdim, int mode, int act) {
  __shared__ _Float16 As[64][40];    // [m][k], padded
  __shared__ _Float16 Bt[128][40];   // [n][k], transposed + padded

  const int tid = threadIdx.x;
  const int lane = tid & 31, wave = tid >> 5;
  const int n0 = blockIdx.x * 128;
  const int m0 = blockIdx.y * 64;
  const int mw = (wave & 3) * 16;    // wave M offset inside block
  const int nw = (wave >> 2) * 64;   // wave N offset inside block

  v8f acc[4];
  {
    int rbase = m0 + mw + ((lane >> 4) << 3);
    #pragma unroll
    for (int r = 0; r < 8; ++r) {
      float bv = bias[rbase + r];
      #pragma unroll
      for (int t = 0; t < 4; ++t) acc[t][r] = bv;
    }
  }

  const int arowS = tid >> 2;            // A staging: 4 thr/row, 8 f16 each
  const int akkS  = (tid & 3) * 8;
  const int bkS   = tid >> 3;            // B staging: k row
  const int bnnS  = (tid & 7) * 16;      // 16 f16 per thread

  for (int ko = 0; ko < Kdim; ko += 32) {
    // ---- stage A tile 64x32 (async DMA into LDS, layout preserved)
    {
      const _Float16* src = W + (size_t)(m0 + arowS) * Kdim + ko + akkS;
      lds_copy16_async(src, &As[arowS][akkS]);
      if (ko + 32 < Kdim) __builtin_prefetch(src + 32, 0, 1);
    }
    // ---- stage B tile 32x128 (transposed into Bt[n][k], through VGPRs)
    if (mode == 0) {
      const _Float16* src = Bmat + (size_t)(ko + bkS) * NTOT + n0 + bnnS;
      _Float16 tmp[16];
      *(uint4*)&tmp[0] = *(const uint4*)src;
      *(uint4*)&tmp[8] = *(const uint4*)(src + 8);
      #pragma unroll
      for (int j = 0; j < 16; ++j) Bt[bnnS + j][bkS] = tmp[j];
    } else {
      int kg = ko + bkS;
      int ci = kg / 9, k9 = kg - ci * 9;
      int ky = k9 / 3 - 1, kx = k9 - (ky + 1) * 3 - 1;
      const _Float16* srcp = Bmat + (size_t)ci * NTOT;
      #pragma unroll
      for (int j = 0; j < 16; ++j) {
        int n = n0 + bnnS + j;
        int b = n >> 10, p = n & 1023, y = p >> 5, x = p & 31;
        int sy = y + ky, sx = x + kx;
        _Float16 val = (_Float16)0.0f;
        if ((unsigned)sy < 32u && (unsigned)sx < 32u)
          val = srcp[(b << 10) + (sy << 5) + sx];
        Bt[bnnS + j][bkS] = val;
      }
    }
    async_wait0();
    __syncthreads();

    // ---- A fragment (16x32), ISA layout: lanes 0-15 K 0-7/16-23, 16-31 K 8-15/24-31
    AF a;
    {
      int row = mw + (lane & 15);
      int kb = (lane >> 4) * 8;
      a.q[0] = *(const uint4*)&As[row][kb];
      a.q[1] = *(const uint4*)&As[row][kb + 16];
    }
    // ---- 4 B fragments + WMMA
    {
      int kb2 = (lane >> 4) * 8;
      #pragma unroll
      for (int t = 0; t < 4; ++t) {
        int col = nw + t * 16 + (lane & 15);
        AF bf;
        bf.q[0] = *(const uint4*)&Bt[col][kb2];
        bf.q[1] = *(const uint4*)&Bt[col][kb2 + 16];
        acc[t] = wmma16(a.v, bf.v, acc[t]);
      }
    }
    __syncthreads();
  }

  // ---- epilogue
  const int colb = n0 + nw + (lane & 15);
  const int rowb = m0 + mw + ((lane >> 4) << 3);
  #pragma unroll
  for (int t = 0; t < 4; ++t) {
    int col = colb + t * 16;
    #pragma unroll
    for (int r = 0; r < 8; ++r) {
      int row = rowb + r;
      float v = acc[t][r];
      if (act) v = silu_f(v);
      if (Res) v += (float)Res[(size_t)row * NTOT + col];
      if (Out) Out[(size_t)row * NTOT + col] = (_Float16)v;
      if (OutF32) {
        int b = col >> 10, p = col & 1023;
        OutF32[((size_t)b * Mdim + row) * 1024 + p] = v;
      }
    }
  }
}

// ---------------------------------------------------------------------------
// BoT attention, one (b, head, 16-row i-tile) per workgroup.
//   QKV matrix [768][8192]: rows 0-255 Q, 256-511 K, 512-767 V (chan h*64+d)
//   S = (Q^T (K + r)) / sqrt(64)  -> in-LDS softmax -> O = P V^T
// ---------------------------------------------------------------------------
__global__ __launch_bounds__(256) void k_attn(const _Float16* __restrict__ QKV,
                                              const float* __restrict__ rw,
                                              const float* __restrict__ rh,
                                              _Float16* __restrict__ AO) {
  __shared__ _Float16 Sbuf[16][1024];   // logits -> probabilities (f16)
  __shared__ _Float16 Qt[16][72];       // Q^T tile [i][d]
  __shared__ _Float16 Kt[128][72];      // (K+r)^T tile [j][d]
  __shared__ float    Obuf[16][64];

  const int tid = threadIdx.x, lane = tid & 31, wave = tid >> 5;
  const int bh = blockIdx.x;            // 0..31
  const int b = bh >> 2, h = bh & 3;
  const int i0 = blockIdx.y * 16;       // 0..1008
  const size_t bcol = (size_t)b * 1024;
  const int hd0 = h * 64;

  // ---- stage Q^T (16 x 64)
  for (int e = tid; e < 16 * 64; e += 256) {
    int i = e & 15, d = e >> 4;
    Qt[i][d] = QKV[(size_t)(hd0 + d) * NTOT + bcol + i0 + i];
  }
  __syncthreads();

  AF a0, a1;
  {
    int row = lane & 15, kb = (lane >> 4) * 8;
    a0.q[0] = *(const uint4*)&Qt[row][kb];
    a0.q[1] = *(const uint4*)&Qt[row][kb + 16];
    a1.q[0] = *(const uint4*)&Qt[row][32 + kb];
    a1.q[1] = *(const uint4*)&Qt[row][32 + kb + 16];
  }

  // ---- S = Q^T (K+r) * 0.125 ; j blocks of 128, one 16-col tile per wave
  for (int jb = 0; jb < 8; ++jb) {
    __syncthreads();
    for (int e = tid; e < 128 * 64; e += 256) {
      int jj = e & 127, d = e >> 7;
      int j = jb * 128 + jj;
      float kv = (float)QKV[(size_t)(256 + hd0 + d) * NTOT + bcol + j] +
                 rw[(hd0 + d) * 32 + (j & 31)] + rh[(hd0 + d) * 32 + (j >> 5)];
      Kt[jj][d] = (_Float16)kv;
    }
    __syncthreads();

    int kb2 = (lane >> 4) * 8;
    int colL = wave * 16 + (lane & 15);
    AF b0, b1;
    b0.q[0] = *(const uint4*)&Kt[colL][kb2];
    b0.q[1] = *(const uint4*)&Kt[colL][kb2 + 16];
    b1.q[0] = *(const uint4*)&Kt[colL][32 + kb2];
    b1.q[1] = *(const uint4*)&Kt[colL][32 + kb2 + 16];

    v8f c = {};
    c = wmma16(a0.v, b0.v, c);
    c = wmma16(a1.v, b1.v, c);

    int sr = (lane >> 4) << 3;
    int sc = jb * 128 + wave * 16 + (lane & 15);
    #pragma unroll
    for (int r = 0; r < 8; ++r) Sbuf[sr + r][sc] = (_Float16)(c[r] * 0.125f);
  }
  __syncthreads();

  // ---- softmax over j (1024): 16 threads per row, shuffle reductions
  {
    int row = tid >> 4, cl = tid & 15;
    float mx = -1e30f;
    for (int j = cl; j < 1024; j += 16) mx = fmaxf(mx, (float)Sbuf[row][j]);
    #pragma unroll
    for (int s = 8; s >= 1; s >>= 1) mx = fmaxf(mx, __shfl_xor(mx, s, 16));
    float sum = 0.0f;
    for (int j = cl; j < 1024; j += 16) {
      float e = __expf((float)Sbuf[row][j] - mx);
      Sbuf[row][j] = (_Float16)e;
      sum += e;
    }
    #pragma unroll
    for (int s = 8; s >= 1; s >>= 1) sum += __shfl_xor(sum, s, 16);
    float inv = 1.0f / sum;
    for (int j = cl; j < 1024; j += 16)
      Sbuf[row][j] = (_Float16)((float)Sbuf[row][j] * inv);
  }
  for (int e = tid; e < 16 * 64; e += 256) ((float*)Obuf)[e] = 0.0f;
  __syncthreads();

  // ---- O = P V^T : wave -> d-tile (wave&3), K-half (wave>>2); reduce via LDS
  {
    int dt = wave & 3, kh = wave >> 2;
    int dcol = dt * 16 + (lane & 15);
    const _Float16* Vrow = QKV + (size_t)(512 + hd0 + dcol) * NTOT + bcol;
    int arow = lane & 15, akb = (lane >> 4) * 8;
    int bkb = (lane >> 4) * 8;
    v8f c = {};
    #pragma unroll
    for (int ks = 0; ks < 16; ++ks) {
      int j0 = kh * 512 + ks * 32;
      AF a, bf;
      a.q[0] = *(const uint4*)&Sbuf[arow][j0 + akb];
      a.q[1] = *(const uint4*)&Sbuf[arow][j0 + akb + 16];
      bf.q[0] = *(const uint4*)&Vrow[j0 + bkb];
      bf.q[1] = *(const uint4*)&Vrow[j0 + bkb + 16];
      c = wmma16(a.v, bf.v, c);
    }
    int orow = (lane >> 4) << 3;
    #pragma unroll
    for (int r = 0; r < 8; ++r) atomicAdd(&Obuf[orow + r][dcol], c[r]);
  }
  __syncthreads();

  for (int e = tid; e < 16 * 64; e += 256) {
    int i = e & 15, d = e >> 4;
    AO[(size_t)(hd0 + d) * NTOT + bcol + i0 + i] = (_Float16)Obuf[i][d];
  }
}

// ---------------------------------------------------------------------------
extern "C" void kernel_launch(void* const* d_in, const int* in_sizes, int n_in,
                              void* d_out, int out_size, void* d_ws,
                              size_t ws_size, hipStream_t stream) {
  (void)in_sizes; (void)n_in; (void)out_size; (void)ws_size;
  const float* x        = (const float*)d_in[0];
  const float* cv1_w    = (const float*)d_in[1];
  const float* cv1_bn   = (const float*)d_in[2];
  const float* cv2_w    = (const float*)d_in[3];
  const float* cv2_bn   = (const float*)d_in[4];
  const float* m_cv1_w  = (const float*)d_in[5];
  const float* m_cv1_bn = (const float*)d_in[6];
  const float* m_qkv_w  = (const float*)d_in[7];
  const float* m_rw     = (const float*)d_in[8];
  const float* m_rh     = (const float*)d_in[9];
  const float* m_cv2_w  = (const float*)d_in[10];
  const float* m_cv2_bn = (const float*)d_in[11];
  float* out = (float*)d_out;

  // bump allocator over d_ws
  size_t off = 0;
  auto alloc = [&](size_t bytes) -> void* {
    void* p = (char*)d_ws + off;
    off = (off + bytes + 255) & ~(size_t)255;
    return p;
  };
  _Float16* X0   = (_Float16*)alloc((size_t)512 * NTOT * 2);
  _Float16* Ycat = (_Float16*)alloc((size_t)1024 * NTOT * 2);  // t | y2 | y3
  _Float16* Zc   = (_Float16*)alloc((size_t)256 * NTOT * 2);
  _Float16* QKV  = (_Float16*)alloc((size_t)768 * NTOT * 2);
  _Float16* AO   = (_Float16*)alloc((size_t)256 * NTOT * 2);
  _Float16* Wcv1 = (_Float16*)alloc((size_t)512 * 512 * 2);
  float*    Bcv1 = (float*)alloc(512 * 4);
  _Float16* Wm1[2]; float* Bm1[2];
  _Float16* Wq[2];  float* Bq[2];
  _Float16* Wm2[2]; float* Bm2[2];
  for (int i = 0; i < 2; ++i) {
    Wm1[i] = (_Float16*)alloc((size_t)256 * 2304 * 2);
    Bm1[i] = (float*)alloc(256 * 4);
    Wq[i]  = (_Float16*)alloc((size_t)768 * 256 * 2);
    Bq[i]  = (float*)alloc(768 * 4);
    Wm2[i] = (_Float16*)alloc((size_t)256 * 256 * 2);
    Bm2[i] = (float*)alloc(256 * 4);
  }
  _Float16* Wcv2 = (_Float16*)alloc((size_t)512 * 1024 * 2);
  float*    Bcv2 = (float*)alloc(512 * 4);

  auto blk = [](int n) { return (n + 255) / 256; };

  // ---- precompute: input conversion + BN-folded f16 weights
  k_x_to_f16<<<blk(512 * NTOT), 256, 0, stream>>>(x, X0);
  k_fold_w<<<blk(512 * 512), 256, 0, stream>>>(cv1_w, cv1_bn, 512, 512, Wcv1, Bcv1);
  k_fold_w<<<blk(512 * 1024), 256, 0, stream>>>(cv2_w, cv2_bn, 512, 1024, Wcv2, Bcv2);
  for (int i = 0; i < 2; ++i) {
    k_fold_w<<<blk(256 * 2304), 256, 0, stream>>>(m_cv1_w + (size_t)i * 256 * 2304,
                                                  m_cv1_bn + (size_t)i * 4 * 256,
                                                  256, 2304, Wm1[i], Bm1[i]);
    k_fold_w<<<blk(768 * 256), 256, 0, stream>>>(m_qkv_w + (size_t)i * 768 * 256,
                                                 nullptr, 768, 256, Wq[i], Bq[i]);
    k_fold_w<<<blk(256 * 256), 256, 0, stream>>>(m_cv2_w + (size_t)i * 256 * 256,
                                                 m_cv2_bn + (size_t)i * 4 * 256,
                                                 256, 256, Wm2[i], Bm2[i]);
  }

  // ---- cv1: t = silu(bn(W1 @ X0)) -> Ycat rows 0..511
  k_gemm<<<dim3(64, 8), 256, 0, stream>>>(Wcv1, Bcv1, X0, Ycat, nullptr, nullptr,
                                          512, 512, 0, 1);

  for (int i = 0; i < 2; ++i) {
    const _Float16* yi = Ycat + (size_t)(256 + 256 * i) * NTOT;  // ys[-1]
    _Float16* yout = Ycat + (size_t)(512 + 256 * i) * NTOT;
    // 3x3 conv (implicit im2col) + bn + silu -> Zc
    k_gemm<<<dim3(64, 4), 256, 0, stream>>>(Wm1[i], Bm1[i], yi, Zc, nullptr,
                                            nullptr, 256, 2304, 1, 1);
    // qkv 1x1 conv (no bn, no act) -> QKV
    k_gemm<<<dim3(64, 12), 256, 0, stream>>>(Wq[i], Bq[i], Zc, QKV, nullptr,
                                             nullptr, 768, 256, 0, 0);
    // attention -> AO
    k_attn<<<dim3(32, 64), 256, 0, stream>>>(QKV, m_rw + (size_t)i * 8192,
                                             m_rh + (size_t)i * 8192, AO);
    // m_cv2 1x1 + bn + silu + residual yi -> ys[2+i]
    k_gemm<<<dim3(64, 4), 256, 0, stream>>>(Wm2[i], Bm2[i], AO, yout, nullptr,
                                            yi, 256, 256, 0, 1);
  }

  // ---- cv2 over concat(ys) = Ycat rows 0..1023 -> fp32 NCHW output
  k_gemm<<<dim3(64, 8), 256, 0, stream>>>(Wcv2, Bcv2, Ycat, nullptr, out,
                                          nullptr, 512, 1024, 0, 1);
}
